// ScatterNet_61744449848108
// MI455X (gfx1250) — matrix operand order repaired and checked by reading, not compiled, not run on hardware
//
#include <hip/hip_runtime.h>
#include <hip/hip_bf16.h>
#include <stdint.h>

typedef __attribute__((ext_vector_type(16))) _Float16 v16h;
typedef __attribute__((ext_vector_type(8)))  _Float16 v8h;
typedef __attribute__((ext_vector_type(8)))  float    v8f;

#define LEAKY_SLOPE 0.01f
__device__ __forceinline__ float leaky(float v){ return v > 0.0f ? v : LEAKY_SLOPE * v; }

// ---------------- graph / diffusion kernels ----------------

__global__ void k_init_deg(float* deg, int n){
  int i = blockIdx.x*blockDim.x + threadIdx.x;
  if (i < n) deg[i] = 1.0f;                      // self-loop contributes 1
}

__global__ void k_deg_accum(const int* __restrict__ dst, float* __restrict__ deg, int E){
  int e = blockIdx.x*blockDim.x + threadIdx.x;
  if (e < E) atomicAdd(&deg[dst[e]], 1.0f);
}

__global__ void k_dinv(float* deg, int n){
  int i = blockIdx.x*blockDim.x + threadIdx.x;
  if (i < n) deg[i] = rsqrtf(deg[i]);            // deg >= 1 always
}

__global__ void k_edge_w(const int* __restrict__ src, const int* __restrict__ dst,
                         const float* __restrict__ dinv, float* __restrict__ w, int E){
  int e = blockIdx.x*blockDim.x + threadIdx.x;
  if (e < E) w[e] = dinv[src[e]] * dinv[dst[e]];
}

__global__ void k_copy(const float* __restrict__ a, float* __restrict__ b, int m){
  int i = blockIdx.x*blockDim.x + threadIdx.x;
  if (i < m) b[i] = a[i];
}

template<int C>
__global__ void k_edge_scatter(const int* __restrict__ src, const int* __restrict__ dst,
                               const float* __restrict__ w, const float* __restrict__ cur,
                               float* __restrict__ agg, int E){
  int e = blockIdx.x*blockDim.x + threadIdx.x;
  if (e >= E) return;
  int s = src[e], d = dst[e];
  float wv = w[e];
  const float* xs = cur + (size_t)s * C;
  float*       ad = agg + (size_t)d * C;
#pragma unroll
  for (int c = 0; c < C; ++c) atomicAdd(ad + c, wv * xs[c]);
}

// next = 0.5*(cur + agg + dinv^2*cur); re-zero agg; optional snapshot
template<int C>
__global__ void k_combine(float* __restrict__ cur, float* __restrict__ agg,
                          const float* __restrict__ dinv, float* __restrict__ snap, int n){
  int idx = blockIdx.x*blockDim.x + threadIdx.x;
  if (idx >= n * C) return;
  int node = idx / C;
  float di = dinv[node];
  float v = 0.5f * (cur[idx] + agg[idx] + di * di * cur[idx]);
  cur[idx] = v;
  agg[idx] = 0.0f;
  if (snap) snap[idx] = v;
}

// s1[n,c,j] = |snap_j[n,c] - snap_{j+1}[n,c]|   (phase-1 snapshots, C=3 layout)
__global__ void k_s1(const float* __restrict__ snap, size_t snapStride,
                     float* __restrict__ s1, int n){
  int t = blockIdx.x*blockDim.x + threadIdx.x;
  if (t >= n * 3) return;
  int node = t / 3, c = t % 3;
#pragma unroll
  for (int j = 0; j < 4; ++j){
    const float* a = snap + (size_t)j * snapStride + (size_t)node * 3 + c;
    s1[(size_t)node * 12 + c * 4 + j] = fabsf(a[0] - a[snapStride]);
  }
}

// Assemble 33 features per node (exact reference layout).
__global__ void k_feats(const float* __restrict__ x, const float* __restrict__ s1,
                        const float* __restrict__ snap2, size_t snapStride,
                        float* __restrict__ feats, int n){
  int t = blockIdx.x*blockDim.x + threadIdx.x;
  if (t >= n * 3) return;
  int node = t / 3, c = t % 3;
  float* out = feats + (size_t)node * 33;
  out[c] = x[(size_t)node * 3 + c];                                  // row 0: s0
#pragma unroll
  for (int j = 0; j < 4; ++j)                                        // rows 1..4: s1
    out[(1 + j) * 3 + c] = s1[(size_t)node * 12 + c * 4 + j];
  const int FENG[6] = {4, 8, 9, 12, 13, 14};
#pragma unroll
  for (int k = 0; k < 6; ++k){                                       // rows 5..10: s2
    int idx = c * 16 + FENG[k];
    int wv = idx / 12, rem = idx % 12;
    int c2 = rem / 4,  f  = rem % 4;
    const float* a = snap2 + (size_t)wv * snapStride + (size_t)node * 12 + c2 * 4 + f;
    out[(5 + k) * 3 + c] = fabsf(a[0] - a[snapStride]);
  }
}

// ---------------- scatter moments ----------------

__global__ void k_count(const int* __restrict__ batch, float* __restrict__ gcnt, int n){
  int i = blockIdx.x*blockDim.x + threadIdx.x;
  if (i < n) atomicAdd(&gcnt[batch[i]], 1.0f);
}
__global__ void k_gsum(const int* __restrict__ batch, const float* __restrict__ feats,
                       float* __restrict__ gsum, int n){
  int t = blockIdx.x*blockDim.x + threadIdx.x;
  if (t >= n * 33) return;
  int node = t / 33, f = t % 33;
  atomicAdd(&gsum[(size_t)batch[node] * 33 + f], feats[t]);
}
__global__ void k_mean(float* __restrict__ gsum, const float* __restrict__ gcnt, int G){
  int t = blockIdx.x*blockDim.x + threadIdx.x;
  if (t < G * 33) gsum[t] /= fmaxf(gcnt[t / 33], 1.0f);
}
__global__ void k_dev(const int* __restrict__ batch, const float* __restrict__ feats,
                      const float* __restrict__ gmean, float* __restrict__ gvar,
                      float* __restrict__ gskew, int n){
  int t = blockIdx.x*blockDim.x + threadIdx.x;
  if (t >= n * 33) return;
  int node = t / 33, f = t % 33;
  size_t gi = (size_t)batch[node] * 33 + f;
  float d = feats[t] - gmean[gi];
  atomicAdd(&gvar[gi],  d * d);
  atomicAdd(&gskew[gi], d * d * d);
}
// h = leaky(concat[mean, var, skew]) -> (G, 99) f32
__global__ void k_hmom(const float* __restrict__ gmean, const float* __restrict__ gvar,
                       const float* __restrict__ gskew, const float* __restrict__ gcnt,
                       float* __restrict__ hmom, int G){
  int t = blockIdx.x*blockDim.x + threadIdx.x;
  if (t >= G * 33) return;
  int g = t / 33, f = t % 33;
  float cnt = fmaxf(gcnt[g], 1.0f);
  hmom[(size_t)g * 99 + f     ] = leaky(gmean[t]);
  hmom[(size_t)g * 99 + 33 + f] = leaky(gvar[t] / cnt);
  hmom[(size_t)g * 99 + 66 + f] = leaky(gskew[t] / cnt);
}

// f32 (G,99) -> f16 (G,128) zero-padded in K
__global__ void k_a0(const float* __restrict__ hmom, _Float16* __restrict__ A0, int G){
  int t = blockIdx.x*blockDim.x + threadIdx.x;
  if (t >= G * 128) return;
  int g = t / 128, k = t % 128;
  A0[t] = (_Float16)(k < 99 ? hmom[(size_t)g * 99 + k] : 0.0f);
}

// f32 (K,N) row-major -> f16 TRANSPOSED (Np,Kp) zero-padded: Bt[n][k] = B[k][n].
// Transposed staging makes each lane's WMMA B-fragment a single contiguous
// 32-byte v16h load (k order used here, same as A: lanes 0-15 hold K=0..15 of col N=lane,
// lanes 16-31 hold K=16..31).
__global__ void k_padwT(const float* __restrict__ src, int K, int N,
                        _Float16* __restrict__ dst, int Kp, int Np){
  int t = blockIdx.x*blockDim.x + threadIdx.x;
  if (t >= Np * Kp) return;
  int nn = t / Kp, k = t % Kp;
  dst[t] = (k < K && nn < N) ? (_Float16)src[(size_t)k * N + nn] : (_Float16)0.0f;
}

// ---------------- WMMA GEMM (one wave -> one 16x16 C tile) ----------------
// C = act(A[M,KP]f16 @ Bt[Np,KP]^T f16 + bias), f32 accumulate.
// KP is a template constant so the K-loop fully unrolls into back-to-back
// v_wmma_f32_16x16x32_f16 with hoisted fragment loads.
template<int KP>
__global__ void k_gemm_wmma(const _Float16* __restrict__ A,
                            const _Float16* __restrict__ Bt,
                            const float* __restrict__ bias,
                            float* __restrict__ outF32, int ld32,
                            _Float16* __restrict__ outF16, int ld16,
                            int Nvalid, int actFlag){
  const int lane = threadIdx.x;        // blockDim.x == 32 (one wave)
  const int hs   = lane >> 4;          // half-wave select
  const int l15  = lane & 15;
  const int mrow = blockIdx.x * 16 + l15;
  const int ncol = blockIdx.y * 16 + l15;
  const _Float16* arow = A  + (size_t)mrow * KP;
  const _Float16* brow = Bt + (size_t)ncol * KP;

  v8f acc = {};
#pragma unroll
  for (int k0 = 0; k0 < KP; k0 += 32){
    // A frag: lane<16 holds K = k0+0..15, lanes 16-31 hold K = k0+16..31
    // (same k order as B). Single 32B vector load.
    v16h a = *(const v16h*)(arow + k0 + hs * 16);
    // B frag: lane<16 -> K = k0+0..15 of its column, lane>=16 -> K = k0+16..31.
    // Contiguous in transposed staging: single 32B vector load.
    v16h b = *(const v16h*)(brow + k0 + hs * 16);
    acc = __builtin_amdgcn_wmma_f32_16x16x32_f16(false, a, false, b, (short)0, acc,
                                                 false, false);
  }
  // C/D layout: VGPR r -> row r (lanes 0-15) / row 8+r (lanes 16-31), col = lane&15.
  if (ncol < Nvalid){
    float bv = bias[ncol];
#pragma unroll
    for (int r = 0; r < 8; ++r){
      int row = blockIdx.x * 16 + r + hs * 8;
      float v = acc[r] + bv;
      if (actFlag) v = leaky(v);
      if (outF32) outF32[(size_t)row * ld32 + ncol] = v;
      if (outF16) outF16[(size_t)row * ld16 + ncol] = (_Float16)v;
    }
  }
}

// ---------------- host orchestration ----------------

static inline int snap_slot(int t){
  switch (t){ case 1: return 0; case 2: return 1; case 4: return 2;
              case 8: return 3; case 16: return 4; }
  return -1;
}

extern "C" void kernel_launch(void* const* d_in, const int* in_sizes, int n_in,
                              void* d_out, int out_size, void* d_ws, size_t ws_size,
                              hipStream_t stream){
  const float* x   = (const float*)d_in[0];
  const float* W1  = (const float*)d_in[1];  const float* b1 = (const float*)d_in[2];
  const float* W2  = (const float*)d_in[3];  const float* b2 = (const float*)d_in[4];
  const float* W3  = (const float*)d_in[5];  const float* b3 = (const float*)d_in[6];
  const float* We  = (const float*)d_in[7];  const float* be = (const float*)d_in[8];
  const float* Wc  = (const float*)d_in[9];  const float* bc = (const float*)d_in[10];
  const int*   eidx  = (const int*)d_in[11];
  const int*   batch = (const int*)d_in[12];

  const int n = in_sizes[0] / 3;
  const int E = in_sizes[11] / 2;
  const int G = 512;                       // N_GRAPHS (compile-time in reference)
  const int* src = eidx;
  const int* dst = eidx + E;

  // workspace carve-out (256B aligned)
  char* p = (char*)d_ws;
  auto alloc = [&](size_t bytes) -> void* {
    uintptr_t u = ((uintptr_t)p + 255) & ~(uintptr_t)255;
    p = (char*)(u + bytes);
    return (void*)u;
  };
  float* deg   = (float*)alloc((size_t)n * 4);
  float* wedge = (float*)alloc((size_t)E * 4);
  float* cur   = (float*)alloc((size_t)n * 12 * 4);
  float* agg   = (float*)alloc((size_t)n * 12 * 4);
  const size_t snapStride = (size_t)n * 12;
  float* snap  = (float*)alloc(5 * snapStride * 4);
  float* s1    = (float*)alloc((size_t)n * 12 * 4);
  float* feats = (float*)alloc((size_t)n * 33 * 4);
  float* gsum  = (float*)alloc((size_t)G * 33 * 4);
  float* gcnt  = (float*)alloc((size_t)G * 4);
  float* gvar  = (float*)alloc((size_t)G * 33 * 4);
  float* gskew = (float*)alloc((size_t)G * 33 * 4);
  float* hmom  = (float*)alloc((size_t)G * 99 * 4);
  _Float16* A0   = (_Float16*)alloc((size_t)G * 128 * 2);
  _Float16* W1f  = (_Float16*)alloc(128 * 128 * 2);   // transposed (Np,Kp)
  _Float16* W2f  = (_Float16*)alloc(64 * 128 * 2);
  _Float16* W3f  = (_Float16*)alloc(64 * 64 * 2);
  _Float16* Wef  = (_Float16*)alloc(32 * 64 * 2);
  _Float16* Wcf  = (_Float16*)alloc(16 * 32 * 2);
  _Float16* h1   = (_Float16*)alloc((size_t)G * 128 * 2);
  _Float16* h2   = (_Float16*)alloc((size_t)G * 64 * 2);
  _Float16* h3   = (_Float16*)alloc((size_t)G * 64 * 2);
  _Float16* embf = (_Float16*)alloc((size_t)G * 32 * 2);

  const int TB = 256;
  auto blocks = [](long long cnt){ return (unsigned)((cnt + 255) / 256); };

  // ---- GCN normalization ----
  k_init_deg<<<blocks(n), TB, 0, stream>>>(deg, n);
  k_deg_accum<<<blocks(E), TB, 0, stream>>>(dst, deg, E);
  k_dinv<<<blocks(n), TB, 0, stream>>>(deg, n);               // deg -> dinv in place
  k_edge_w<<<blocks(E), TB, 0, stream>>>(src, dst, deg, wedge, E);

  hipMemsetAsync(agg, 0, (size_t)n * 12 * 4, stream);

  // ---- phase 1: 16 diffusions on x (C=3) ----
  k_copy<<<blocks((long long)n * 3), TB, 0, stream>>>(x, cur, n * 3);
  for (int t = 1; t <= 16; ++t){
    k_edge_scatter<3><<<blocks(E), TB, 0, stream>>>(src, dst, wedge, cur, agg, E);
    int sj = snap_slot(t);
    float* sp = (sj >= 0) ? (snap + (size_t)sj * snapStride) : nullptr;
    k_combine<3><<<blocks((long long)n * 3), TB, 0, stream>>>(cur, agg, deg, sp, n);
  }
  k_s1<<<blocks((long long)n * 3), TB, 0, stream>>>(snap, snapStride, s1, n);

  // ---- phase 2: 16 diffusions on s1 (C=12) ----
  k_copy<<<blocks((long long)n * 12), TB, 0, stream>>>(s1, cur, n * 12);
  for (int t = 1; t <= 16; ++t){
    k_edge_scatter<12><<<blocks(E), TB, 0, stream>>>(src, dst, wedge, cur, agg, E);
    int sj = snap_slot(t);
    float* sp = (sj >= 0) ? (snap + (size_t)sj * snapStride) : nullptr;
    k_combine<12><<<blocks((long long)n * 12), TB, 0, stream>>>(cur, agg, deg, sp, n);
  }

  // ---- node features (n,33) ----
  k_feats<<<blocks((long long)n * 3), TB, 0, stream>>>(x, s1, snap, snapStride, feats, n);

  // ---- per-graph moments -> h (G,99), leaky applied ----
  hipMemsetAsync(gsum,  0, (size_t)G * 33 * 4, stream);
  hipMemsetAsync(gcnt,  0, (size_t)G * 4, stream);
  hipMemsetAsync(gvar,  0, (size_t)G * 33 * 4, stream);
  hipMemsetAsync(gskew, 0, (size_t)G * 33 * 4, stream);
  k_count<<<blocks(n), TB, 0, stream>>>(batch, gcnt, n);
  k_gsum<<<blocks((long long)n * 33), TB, 0, stream>>>(batch, feats, gsum, n);
  k_mean<<<blocks((long long)G * 33), TB, 0, stream>>>(gsum, gcnt, G);
  k_dev<<<blocks((long long)n * 33), TB, 0, stream>>>(batch, feats, gsum, gvar, gskew, n);
  k_hmom<<<blocks((long long)G * 33), TB, 0, stream>>>(gsum, gvar, gskew, gcnt, hmom, G);

  // ---- f16 staging for WMMA MLP (weights transposed to (N,Kp)) ----
  k_a0<<<blocks((long long)G * 128), TB, 0, stream>>>(hmom, A0, G);
  k_padwT<<<blocks(128 * 128), TB, 0, stream>>>(W1, 99, 128, W1f, 128, 128);
  k_padwT<<<blocks(64 * 128),  TB, 0, stream>>>(W2, 128, 64, W2f, 128, 64);
  k_padwT<<<blocks(64 * 64),   TB, 0, stream>>>(W3, 64, 64,  W3f, 64, 64);
  k_padwT<<<blocks(32 * 64),   TB, 0, stream>>>(We, 64, 32,  Wef, 64, 32);
  k_padwT<<<blocks(16 * 32),   TB, 0, stream>>>(Wc, 32, 1,   Wcf, 32, 16);

  float* out_emb = (float*)d_out;                   // (512,32)
  float* out_val = (float*)d_out + (size_t)G * 32;  // (512,1)

  dim3 wave(32, 1, 1);
  // L1: (512,128) = A0(512,128) @ W1, leaky
  k_gemm_wmma<128><<<dim3(G/16, 8), wave, 0, stream>>>(A0, W1f, b1,
                                                       nullptr, 0, h1, 128, 128, 1);
  // L2: (512,64) = h1 @ W2, leaky
  k_gemm_wmma<128><<<dim3(G/16, 4), wave, 0, stream>>>(h1, W2f, b2,
                                                       nullptr, 0, h2, 64, 64, 1);
  // L3: (512,64) = h2 @ W3, no act
  k_gemm_wmma<64><<<dim3(G/16, 4), wave, 0, stream>>>(h2, W3f, b3,
                                                      nullptr, 0, h3, 64, 64, 0);
  // L4: embedding (512,32) = h3 @ We -> d_out (f32) + f16 copy
  k_gemm_wmma<64><<<dim3(G/16, 2), wave, 0, stream>>>(h3, Wef, be,
                                                      out_emb, 32, embf, 32, 32, 0);
  // L5: output (512,1) = embf @ Wc (N padded to 16, only col 0 valid)
  k_gemm_wmma<32><<<dim3(G/16, 1), wave, 0, stream>>>(embf, Wcf, bc,
                                                      out_val, 1, nullptr, 0, 1, 0);
}
